// BinaryConv2d_49761491091791
// MI455X (gfx1250) — matrix operand order repaired and checked by reading, not compiled, not run on hardware
//
#include <hip/hip_runtime.h>

// ---------------------------------------------------------------------------
// Binary-weight 3x3 conv (SAME, stride 1) as implicit GEMM on CDNA5 WMMA.
//   M = N*H*W = 802816 output pixels, N = 128 out channels, K = 3*3*64 = 576
//   A (f16): activations per (ky,kx,ci) K-slice, double-buffered LDS,
//            software-pipelined global loads (issue s+1 before barrier s)
//   B (f16): sign(W) transposed to [co][k], resident in LDS per block
//   C (f32): accumulate in v8f, fused bias + ReLU epilogue
// Block: 256 threads = 8 wave32; tile 64(M) x 128(N); 18 K-steps of 32.
// ---------------------------------------------------------------------------

typedef __attribute__((ext_vector_type(16))) _Float16 v16h;
typedef __attribute__((ext_vector_type(8)))  float    v8f;

#define IH     224
#define IW     224
#define CI     64
#define CO     128
#define KSZ    576        // 3*3*64
#define KSTEPS 18         // 576 / 32
#define MTILE  64         // pixels per block
#define NBLK   12544      // 802816 / 64

#define B_PITCH_H 584     // halves per LDS B row: 576 + 8 pad (16B pitch, conflict-free)
#define A_PITCH_H 40      // halves per LDS A row: 32 + 8 pad  (16B pitch, conflict-free)

struct SMem {
  _Float16 Bw[CO * B_PITCH_H];           // 149,504 B: full 576x128 f16 weights
  _Float16 At[2][MTILE * A_PITCH_H];     //  10,240 B: double-buffered A tile
};                                        // total 159,744 B <= 160 KB

union FragH { v16h v; int4 q[2]; };

// ---------------------------------------------------------------------------
// Kernel 1: Wq[co*576 + k] = sign(W[k*128 + co]) as f16   (k = (ky*3+kx)*64+ci)
// ---------------------------------------------------------------------------
__global__ void quantize_weights(const float* __restrict__ Wsrc,
                                 _Float16* __restrict__ Wq) {
  int i = blockIdx.x * blockDim.x + threadIdx.x;   // i = k*128 + co
  if (i >= KSZ * CO) return;
  int k  = i >> 7;          // /128
  int co = i & 127;
  float w = Wsrc[i];
  float s = (float)(w > 0.0f) - (float)(w < 0.0f); // jnp.sign (sign(0)=0)
  Wq[co * KSZ + k] = (_Float16)s;
}

// ---------------------------------------------------------------------------
// Kernel 2: implicit-GEMM conv. 256 threads (8 wave32) per block.
// Wave (wr,wc): wr in 0..3 -> 16M rows, wc in 0..1 -> 64N cols (4 WMMA tiles).
// ---------------------------------------------------------------------------
__global__ void __launch_bounds__(256)
binconv_wmma(const float* __restrict__ x,
             const _Float16* __restrict__ Wq,
             const float* __restrict__ bias,
             float* __restrict__ out) {
  __shared__ SMem sm;

  const int t    = threadIdx.x;
  const int lane = t & 31;
  const int wave = t >> 5;

  // ---- stage full B (576x128 f16) into LDS via b128 moves ------------------
  {
    // 128 rows x 72 int4 per row = 9216 16-byte transfers
    for (int idx = t; idx < CO * (KSZ / 8); idx += 256) {
      int r = idx / (KSZ / 8);
      int c = idx - r * (KSZ / 8);
      ((int4*)&sm.Bw[r * B_PITCH_H])[c] = ((const int4*)(Wq + r * KSZ))[c];
    }
  }

  // ---- per-thread pixel for cooperative A staging --------------------------
  const long long pbase = (long long)blockIdx.x * MTILE;
  const int m  = t >> 2;              // 0..63 : pixel within tile
  const int c8 = (t & 3) * 8;         // 0/8/16/24 : ci sub-chunk (8 floats)
  long long p = pbase + m;
  int n   = (int)(p / (IH * IW));
  int rem = (int)(p % (IH * IW));
  int ph  = rem / IW;
  int pw  = rem % IW;
  const float* xbase =
      x + ((long long)((n * IH + ph) * IW + pw) * CI + c8);

  // ---- wave tiling ---------------------------------------------------------
  const int wr    = wave >> 1;        // 0..3 : M quarter
  const int wc    = wave & 1;         // 0..1 : N half
  const int mOff  = wr * 16;
  const int nBase = wc * 64;
  const int lmod  = lane & 15;
  const int hi    = lane >> 4;        // 0 or 1 (lane half)

  v8f acc[4];
#pragma unroll
  for (int j = 0; j < 4; ++j) acc[j] = (v8f){};

  // Per-step A gather: 8 consecutive x floats (or zeros at image border).
  auto loadA = [&](int s, float4& f0, float4& f1) {
    int kykx = s >> 1;
    int ci0  = (s & 1) * 32;
    int ky   = kykx / 3;
    int kx   = kykx - ky * 3;
    int ih   = ph + ky - 1;
    int iw   = pw + kx - 1;
    f0 = make_float4(0.f, 0.f, 0.f, 0.f);
    f1 = f0;
    if ((unsigned)ih < (unsigned)IH && (unsigned)iw < (unsigned)IW) {
      const float* px = xbase + ((ky - 1) * IW + (kx - 1)) * CI + ci0;
      f0 = *(const float4*)(px);
      f1 = *(const float4*)(px + 4);
    }
  };

  // ---- K loop, software-pipelined: load(s+1) issued before barrier(s) ------
  float4 n0, n1;
  loadA(0, n0, n1);

#pragma unroll 2
  for (int s = 0; s < KSTEPS; ++s) {
    const int buf = s & 1;

    // convert + store current A slice (waits on loads issued last iteration)
    alignas(16) _Float16 h8[8] = {
        (_Float16)n0.x, (_Float16)n0.y, (_Float16)n0.z, (_Float16)n0.w,
        (_Float16)n1.x, (_Float16)n1.y, (_Float16)n1.z, (_Float16)n1.w};
    *(int4*)&sm.At[buf][m * A_PITCH_H + c8] = *(const int4*)h8;

    // issue next step's global loads; wait deferred to next iteration
    float4 p0 = make_float4(0.f, 0.f, 0.f, 0.f), p1 = p0;
    if (s + 1 < KSTEPS) loadA(s + 1, p0, p1);

    __syncthreads();   // A(buf) ready; also covers B staging on s==0

    // A fragment (16x32 f16 layout): lanes 0-15 -> K {0..7,16..23},
    // lanes 16-31 -> K {8..15,24..31}; two 16B chunks at +0/+32 (or +16/+48).
    FragH a;
    const char* abase =
        (const char*)&sm.At[buf][(mOff + lmod) * A_PITCH_H] + hi * 16;
    a.q[0] = *(const int4*)(abase);
    a.q[1] = *(const int4*)(abase + 32);

    const int kb = s * 64 + hi * 16;  // byte offset in B row: k0*2 + lane-half*8 halves
#pragma unroll
    for (int j = 0; j < 4; ++j) {
      // B fragment (32x16 f16): lane -> column N, lane-half -> K half.
      FragH b;
      const char* bb =
          (const char*)&sm.Bw[(nBase + j * 16 + lmod) * B_PITCH_H] + kb;
      b.q[0] = *(const int4*)(bb);
      b.q[1] = *(const int4*)(bb + 32);
      acc[j] = __builtin_amdgcn_wmma_f32_16x16x32_f16(
          /*neg_a=*/false, a.v, /*neg_b=*/false, b.v,
          /*c_mod=*/(short)0, acc[j], /*reuse_a=*/false, /*reuse_b=*/false);
    }

    n0 = p0;
    n1 = p1;
  }

  // ---- epilogue: bias + ReLU, scatter per documented C/D layout ------------
#pragma unroll
  for (int j = 0; j < 4; ++j) {
    const int col = nBase + j * 16 + lmod;
    const float bv = bias[col];
    v8f c = acc[j];
#pragma unroll
    for (int r = 0; r < 8; ++r) {
      long long row = pbase + mOff + r + hi * 8;
      float v = c[r] + bv;
      out[row * CO + col] = v > 0.0f ? v : 0.0f;
    }
  }
}

// ---------------------------------------------------------------------------
extern "C" void kernel_launch(void* const* d_in, const int* in_sizes, int n_in,
                              void* d_out, int out_size, void* d_ws,
                              size_t ws_size, hipStream_t stream) {
  const float* x    = (const float*)d_in[0];   // (16,224,224,64)  fp32 NHWC
  const float* Wsrc = (const float*)d_in[1];   // (3,3,64,128)     fp32 HWIO
  const float* bias = (const float*)d_in[2];   // (128,)           fp32
  float* out        = (float*)d_out;           // (16,224,224,128) fp32
  _Float16* Wq      = (_Float16*)d_ws;         // 73728 f16 = 144 KB scratch

  (void)in_sizes; (void)n_in; (void)out_size; (void)ws_size;

  quantize_weights<<<(KSZ * CO + 255) / 256, 256, 0, stream>>>(Wsrc, Wq);

  // 802816 pixels / 64 per block = 12544 blocks, 256 threads (8 wave32)
  binconv_wmma<<<NBLK, 256, 0, stream>>>(x, Wq, bias, out);
}
